// ChebGraphConv_5463198400727
// MI455X (gfx1250) — matrix operand order repaired and checked by reading, not compiled, not run on hardware
//
#include <hip/hip_runtime.h>

// ---------------------------------------------------------------------------
// ChebGraphConv (K=3) for MI455X / gfx1250, wave32 + WMMA + async-to-LDS.
//
//   prep:  Abf = bf16(A_norm)    (A reused 192x -> convert once, halve traffic)
//   AX  = A @ X    (per bt)      bf16 WMMA 16x16x32, f32 accumulate
//   AAX = A @ AX   (per bt)      bf16 WMMA 16x16x32, f32 accumulate
//   y   = X@(W0-W2) + AX@(-W1) + AAX@(2*W2) + bias     f32 WMMA 16x16x4
//
// A tiles move global->LDS via GLOBAL_LOAD_ASYNC_TO_LDS_B128 (ASYNCcnt),
// double-buffered so the async engine fills tile kt+1 while WMMAs chew on kt.
// ---------------------------------------------------------------------------

typedef __attribute__((ext_vector_type(16))) __bf16 v16bf;
typedef __attribute__((ext_vector_type(8)))  __bf16 v8bf;
typedef __attribute__((ext_vector_type(8)))  float  v8f;
typedef __attribute__((ext_vector_type(2)))  float  v2f;
typedef __attribute__((ext_vector_type(4)))  int    v4i;

// address-space qualified pointer types for the async-to-LDS builtin
typedef __attribute__((address_space(1))) v4i* gas_v4i_p;   // global
typedef __attribute__((address_space(3))) v4i* las_v4i_p;   // LDS

#define N_NODES 2048
#define CCH     64
#define MT      128   // M rows per workgroup
#define KT      64    // K depth staged per iteration
#define LDSPAD  80    // padded LDS row stride (bf16 elems): 160B rows, 16B aligned

union AB16 { v16bf v; v8bf h[2]; };

#if defined(__has_builtin)
#if __has_builtin(__builtin_amdgcn_global_load_async_to_lds_b128)
#define HAS_ASYNC_LDS 1
#endif
#endif

__device__ __forceinline__ void async_copy16(const __bf16* g, __bf16* l)
{
#ifdef HAS_ASYNC_LDS
    __builtin_amdgcn_global_load_async_to_lds_b128(
        (gas_v4i_p)(void*)g, (las_v4i_p)(void*)l, 0, 0);
#else
    *reinterpret_cast<uint4*>(l) = *reinterpret_cast<const uint4*>(g);
#endif
}

__device__ __forceinline__ void async_wait0()
{
#ifdef HAS_ASYNC_LDS
#if __has_builtin(__builtin_amdgcn_s_wait_asynccnt)
    __builtin_amdgcn_s_wait_asynccnt(0);
#else
    asm volatile("s_wait_asynccnt 0x0" ::: "memory");
#endif
#endif
}

// --------------------------- weight combine --------------------------------
__global__ void cheb_combine_w(const float* __restrict__ w, float* __restrict__ wc)
{
    int t = blockIdx.x * blockDim.x + threadIdx.x;   // 0..4095 over 64x64
    if (t < 64 * 64) {
        wc[t]            = w[t] - w[2 * 4096 + t];   // W0 - W2
        wc[4096 + t]     = -w[4096 + t];             // -W1
        wc[2 * 4096 + t] = 2.0f * w[2 * 4096 + t];   // 2*W2
    }
}

// --------------------------- A f32 -> bf16 ---------------------------------
__global__ __launch_bounds__(256)
void cheb_convA(const float* __restrict__ A, __bf16* __restrict__ Abf)
{
    long t = ((long)blockIdx.x * 256 + threadIdx.x) * 4;   // 4 elems/thread
    float4 v = *reinterpret_cast<const float4*>(A + t);
    union { __bf16 h[4]; uint2 u; } pk;
    pk.h[0] = (__bf16)v.x; pk.h[1] = (__bf16)v.y;
    pk.h[2] = (__bf16)v.z; pk.h[3] = (__bf16)v.w;
    *reinterpret_cast<uint2*>(Abf + t) = pk.u;
}

// --------------------- A(2048x2048) @ B(2048x64) per bt --------------------
// Abf: bf16 row-major [N][N].  B/Out: f32 [96][N][64].
__global__ __launch_bounds__(256)
void cheb_spmm_bf16(const __bf16* __restrict__ Abf,
                    const float* __restrict__ B,
                    float* __restrict__ Out)
{
    __shared__ __bf16 As[2][MT * LDSPAD];    // [m][k]  2 x 20 KB
    __shared__ __bf16 Bs[2][CCH * LDSPAD];   // [c][k]  2 x 10 KB (transposed)

    const int tid  = threadIdx.x;
    const int wave = tid >> 5;
    const int lane = tid & 31;
    const int l16  = lane & 15;
    const int hi   = lane >> 4;

    const int  m0     = blockIdx.x * MT;
    const int  bt     = blockIdx.y;
    const long btbase = (long)bt * N_NODES * CCH;

    v8f acc[4];
    #pragma unroll
    for (int j = 0; j < 4; ++j)
        acc[j] = (v8f){0.f, 0.f, 0.f, 0.f, 0.f, 0.f, 0.f, 0.f};

    // ---- tile staging: A via async-to-LDS (bf16), B via convert+transpose ----
    auto stage = [&](int kt, int p) {
        // A tile MT x KT bf16 = 1024 x 16B chunks; 4 chunks/thread
        #pragma unroll
        for (int i = 0; i < 4; ++i) {
            int chunk = i * 256 + tid;
            int row   = chunk >> 3;          // /8 chunks per row
            int kc    = (chunk & 7) * 8;     // bf16 elems
            async_copy16(&Abf[(long)(m0 + row) * N_NODES + kt + kc],
                         &As[p][row * LDSPAD + kc]);
        }
        // B tile KT x 64, f32 -> bf16, transposed: Bs[c][k] = B[kt+k][c]
        #pragma unroll
        for (int i = 0; i < 4; ++i) {
            int flat = (i * 256 + tid) * 4;
            int k    = flat >> 6;
            int c    = flat & 63;
            float4 v = *reinterpret_cast<const float4*>(
                &B[btbase + (long)(kt + k) * CCH + c]);
            Bs[p][(c + 0) * LDSPAD + k] = (__bf16)v.x;
            Bs[p][(c + 1) * LDSPAD + k] = (__bf16)v.y;
            Bs[p][(c + 2) * LDSPAD + k] = (__bf16)v.z;
            Bs[p][(c + 3) * LDSPAD + k] = (__bf16)v.w;
        }
    };

    auto compute = [&](int p) {
        #pragma unroll
        for (int ks = 0; ks < KT; ks += 32) {
            // A frag 16x32 bf16: lanes<16 K={0..7,16..23}, lanes>=16 +8
            AB16 a;
            const int m  = wave * 16 + l16;
            const int kf = ks + hi * 8;
            a.h[0] = *reinterpret_cast<const v8bf*>(&As[p][m * LDSPAD + kf]);
            a.h[1] = *reinterpret_cast<const v8bf*>(&As[p][m * LDSPAD + kf + 16]);
            #pragma unroll
            for (int j = 0; j < 4; ++j) {
                // B frag 32x16 bf16: lanes<16 K={0..7,16..23} of col N=l16, lanes>=16 +8
                AB16 b;
                const int cc = j * 16 + l16;
                const int kb = ks + hi * 8;
                b.h[0] = *reinterpret_cast<const v8bf*>(&Bs[p][cc * LDSPAD + kb]);
                b.h[1] = *reinterpret_cast<const v8bf*>(&Bs[p][cc * LDSPAD + kb + 16]);
                acc[j] = __builtin_amdgcn_wmma_f32_16x16x32_bf16(
                    false, a.v, false, b.v, (short)0, acc[j], false, false);
            }
        }
    };

    const int NKT = N_NODES / KT;   // 32
    stage(0, 0);
    async_wait0();
    __syncthreads();

    for (int kt = 0; kt < NKT; ++kt) {
        const int p = kt & 1;
        if (kt + 1 < NKT) stage((kt + 1) * KT, p ^ 1);   // prefetch next tile
        compute(p);                                      // overlaps async fills
        async_wait0();
        __syncthreads();
    }

    // C/D layout: VGPR r, lanes<16 -> (M=r, N=l16); lanes>=16 -> (M=r+8, N=l16)
    const int mo = m0 + wave * 16 + hi * 8;
    #pragma unroll
    for (int j = 0; j < 4; ++j)
        #pragma unroll
        for (int r = 0; r < 8; ++r)
            Out[btbase + (long)(mo + r) * CCH + j * 16 + l16] = acc[j][r];
}

// ------------------- projection: y = X@Wc0 + AX@Wc1 + AAX@Wc2 + b ----------
__global__ __launch_bounds__(256)
void cheb_proj(const float* __restrict__ X,
               const float* __restrict__ AX,
               const float* __restrict__ AAX,
               const float* __restrict__ Wc,    // [3][64][64] f32
               const float* __restrict__ bias,  // [64]
               float* __restrict__ Y)
{
    __shared__ float Wl[3 * 64 * 64];   // 48 KB
    __shared__ float bl[64];

    const int tid = threadIdx.x;
    for (int i = tid; i < 3 * 64 * 64; i += 256) Wl[i] = Wc[i];
    if (tid < 64) bl[tid] = bias[tid];
    __syncthreads();

    const int wave = tid >> 5;
    const int lane = tid & 31;
    const int l16  = lane & 15;
    const int hi   = lane >> 4;

    const int  bt   = blockIdx.y;
    const int  m0   = blockIdx.x * MT + wave * 16;
    const long base = (long)bt * N_NODES * CCH;

    v8f acc[4];
    #pragma unroll
    for (int j = 0; j < 4; ++j)
        acc[j] = (v8f){0.f, 0.f, 0.f, 0.f, 0.f, 0.f, 0.f, 0.f};

    const float* srcs[3] = {X, AX, AAX};
    #pragma unroll
    for (int s = 0; s < 3; ++s) {
        const float* src = srcs[s];
        const float* wb  = &Wl[s * 4096];
        const long rowbase = base + (long)(m0 + l16) * CCH;
        #pragma unroll 4
        for (int k0 = 0; k0 < 64; k0 += 4) {
            // f32 A frag 16x4: VGPR0 holds K=k0 (lanes<16) / K=k0+2 (lanes>=16)
            float2 f = *reinterpret_cast<const float2*>(&src[rowbase + k0 + 2 * hi]);
            v2f a; a.x = f.x; a.y = f.y;
            #pragma unroll
            for (int j = 0; j < 4; ++j) {
                const int o = j * 16 + l16;
                v2f b;
                b.x = wb[(k0 + 2 * hi) * 64 + o];
                b.y = wb[(k0 + 2 * hi + 1) * 64 + o];
                acc[j] = __builtin_amdgcn_wmma_f32_16x16x4_f32(
                    false, a, false, b, (short)0, acc[j], false, false);
            }
        }
    }

    const int mo = m0 + hi * 8;
    #pragma unroll
    for (int j = 0; j < 4; ++j) {
        const float bv = bl[j * 16 + l16];
        #pragma unroll
        for (int r = 0; r < 8; ++r)
            Y[base + (long)(mo + r) * CCH + j * 16 + l16] = acc[j][r] + bv;
    }
}

// --------------------------------------------------------------------------
extern "C" void kernel_launch(void* const* d_in, const int* in_sizes, int n_in,
                              void* d_out, int out_size, void* d_ws, size_t ws_size,
                              hipStream_t stream)
{
    const float* x    = (const float*)d_in[0];  // [8,12,2048,64]
    const float* A    = (const float*)d_in[1];  // [2048,2048]
    const float* w    = (const float*)d_in[2];  // [3,64,64]
    const float* bias = (const float*)d_in[3];  // [64]
    float*       out  = (float*)d_out;          // [8,12,2048,64]
    float*       ws   = (float*)d_ws;

    const long BTNC = 96L * N_NODES * CCH;      // 12,582,912 floats
    float*  Wc  = ws;                           // 12288 floats (16384 reserved)
    __bf16* Abf = (__bf16*)(ws + 16384);        // 2048*2048 bf16 = 2,097,152 floats
    float*  AX  = ws + 16384 + 2097152;         // BTNC floats
    float*  AAX = AX + BTNC;                    // BTNC floats
    // total ws: (16384 + 2,097,152 + 2*12,582,912)*4 B ~= 104 MB

    cheb_combine_w<<<16, 256, 0, stream>>>(w, Wc);
    cheb_convA<<<4096, 256, 0, stream>>>(A, Abf);   // 2048*2048 / (256*4)

    dim3 grid(N_NODES / MT, 96);                // 16 M-tiles x 96 bt
    cheb_spmm_bf16<<<grid, 256, 0, stream>>>(Abf, x,  AX);
    cheb_spmm_bf16<<<grid, 256, 0, stream>>>(Abf, AX, AAX);
    cheb_proj<<<grid, 256, 0, stream>>>(x, AX, AAX, Wc, bias, out);
}
